// Linformer_9251359555969
// MI455X (gfx1250) — matrix operand order repaired and checked by reading, not compiled, not run on hardware
//
#include <hip/hip_runtime.h>
#include <hip/hip_bf16.h>
#include <stdint.h>

// ---------------------------------------------------------------------------
// Complex Linformer (N=4096, D=512, K=256, H=8, DH=64, DEPTH=2, MULT=4)
// All matmuls as bf16 WMMA (v_wmma_f32_16x16x32_bf16), fp32 accumulation,
// fp32 residual stream. B operands kept transposed ([N][K] row-major).
// LDS staging via GLOBAL_LOAD_ASYNC_TO_LDS_B128 (ASYNCcnt), double-buffered,
// K-step 64.
// ---------------------------------------------------------------------------

typedef __bf16 bf16;
typedef __attribute__((ext_vector_type(16))) __bf16 bf16x16;
typedef __attribute__((ext_vector_type(8)))  float   f32x8;

#define DEPTH  2
#define SEQ    4096
#define DIM    512
#define LRK    256
#define HEADS  8
#define DHEAD  64
#define DFF    2048
#define EPSC   1e-5f
#define SCALEC 0.125f   // DH^-0.5

#define BM 128
#define BN 64
#define BK 64
#define LPA 72   // LDS row pitch: 144B rows, 16B aligned, bank stride 36 (conflict-free)

struct GArgs {
  const bf16* A1; const bf16* B1;      // B* point to Bt ([N][K] row-major)
  const bf16* A2; const bf16* B2;
  long sA1, sB1, sA2, sB2;             // batch strides (elements)
  int  lda1, ldb1, lda2, ldb2;         // row pitches (K-dim pitch)
  int  K1, K2;
  float alpha1, alpha2;
  const float* bias;                   // per-column, nullable
  const float* res;  long sRes;  int ldres;   // fp32 residual, nullable
  float* outF;  long sOutF;  int ldoF;        // fp32 output, nullable
  bf16*  outB;  long sOutB;  int ldoB;        // bf16 output [M][N], nullable
  bf16*  outBT; long sOutBT; int ldoBT;       // bf16 output transposed [N][M], nullable
  int relu;
};

// ---- async global->LDS copies (ASYNCcnt). INST_OFFSET offsets both sides. ----
__device__ __forceinline__ void async_cp64B(bf16* ldst, const bf16* gsrc) {
  unsigned lo = (unsigned)(uintptr_t)ldst;   // LDS byte offset (flat addr low dword)
  asm volatile(
      "global_load_async_to_lds_b128 %0, %1, off\n\t"
      "global_load_async_to_lds_b128 %0, %1, off offset:16\n\t"
      "global_load_async_to_lds_b128 %0, %1, off offset:32\n\t"
      "global_load_async_to_lds_b128 %0, %1, off offset:48"
      :: "v"(lo), "v"(gsrc) : "memory");
}
__device__ __forceinline__ void async_cp32B(bf16* ldst, const bf16* gsrc) {
  unsigned lo = (unsigned)(uintptr_t)ldst;
  asm volatile(
      "global_load_async_to_lds_b128 %0, %1, off\n\t"
      "global_load_async_to_lds_b128 %0, %1, off offset:16"
      :: "v"(lo), "v"(gsrc) : "memory");
}
__device__ __forceinline__ void wait_async0() {
  asm volatile("s_wait_asynccnt 0x0" ::: "memory");
}

__device__ __forceinline__ void stage_tiles(const bf16* __restrict__ Ag, int lda,
                                            const bf16* __restrict__ Bg, int ldb,
                                            int m0, int n0, int kb,
                                            bf16* aD, bf16* bD, int t) {
  // A tile: 128 rows x 64 k. 256 threads x 32 elems (64B) each.
  {
    int row = t >> 1;
    int seg = (t & 1) * 32;
    async_cp64B(aD + row * LPA + seg, Ag + (long)(m0 + row) * lda + kb + seg);
  }
  // Bt tile: 64 rows (n) x 64 k. 256 threads x 16 elems (32B) each.
  {
    int row = t >> 2;
    int seg = (t & 3) * 16;
    async_cp32B(bD + row * LPA + seg, Bg + (long)(n0 + row) * ldb + kb + seg);
  }
}

__device__ __forceinline__ void wmma_step(f32x8 acc[2][2], const bf16* aL, const bf16* bL,
                                          int wm, int wn, int hf, int r, int kk) {
  union V16 { bf16x16 v; uint4 q[2]; };
  V16 af[2], bfm[2];
  // A fragment (16x32): lane group hf -> K {0..7}/{8..15} in q[0], +16 in q[1]
#pragma unroll
  for (int mt = 0; mt < 2; ++mt) {
    const bf16* p = aL + (wm + mt * 16 + r) * LPA + kk;
    af[mt].q[0] = *reinterpret_cast<const uint4*>(p + hf * 8);
    af[mt].q[1] = *reinterpret_cast<const uint4*>(p + 16 + hf * 8);
  }
  // B fragment (32x16): lane = column n, group hf -> K {0..7}/{8..15} in q[0], +16 in q[1]
#pragma unroll
  for (int nt = 0; nt < 2; ++nt) {
    const bf16* p = bL + (wn + nt * 16 + r) * LPA + kk + hf * 8;
    bfm[nt].q[0] = *reinterpret_cast<const uint4*>(p);
    bfm[nt].q[1] = *reinterpret_cast<const uint4*>(p + 16);
  }
#pragma unroll
  for (int mt = 0; mt < 2; ++mt)
#pragma unroll
    for (int nt = 0; nt < 2; ++nt)
      acc[mt][nt] = __builtin_amdgcn_wmma_f32_16x16x32_bf16(
          false, af[mt].v, false, bfm[nt].v, (short)0, acc[mt][nt], false, false);
}

__global__ __launch_bounds__(256) void dual_gemm(GArgs g) {
  __shared__ __align__(16) bf16 aLs[2][BM * LPA];
  __shared__ __align__(16) bf16 bLs[2][BN * LPA];
  const int t    = threadIdx.x;
  const int wave = t >> 5;
  const int lane = t & 31;
  const int hf   = lane >> 4;
  const int r    = lane & 15;
  const int wm   = (wave & 3) * 32;
  const int wn   = (wave >> 2) * 32;
  const int m0   = blockIdx.y * BM;
  const int n0   = blockIdx.x * BN;
  const int z    = blockIdx.z;

  const bf16* A1 = g.A1 + (long)z * g.sA1;
  const bf16* B1 = g.B1 + (long)z * g.sB1;
  const bf16* A2 = g.A2 ? g.A2 + (long)z * g.sA2 : nullptr;
  const bf16* B2 = g.B2 ? g.B2 + (long)z * g.sB2 : nullptr;

  const int it1 = g.K1 >> 6;
  const int it2 = g.K2 >> 6;
  const int itT = it1 + it2;
  const float rs = (it2 > 0) ? g.alpha1 / g.alpha2 : 1.0f;

  auto pick = [&](int i, const bf16*& A, int& la, const bf16*& B, int& lb, int& kb) {
    if (i < it1) { A = A1; la = g.lda1; B = B1; lb = g.ldb1; kb = i << 6; }
    else         { A = A2; la = g.lda2; B = B2; lb = g.ldb2; kb = (i - it1) << 6; }
  };

  f32x8 acc[2][2] = {};

  { const bf16 *A, *B; int la, lb, kb;
    pick(0, A, la, B, lb, kb);
    stage_tiles(A, la, B, lb, m0, n0, kb, aLs[0], bLs[0], t); }
  wait_async0();
  __syncthreads();

  int cur = 0;
  for (int i = 0; i < itT; ++i) {
    if (i + 1 < itT) {
      const bf16 *A, *B; int la, lb, kb;
      pick(i + 1, A, la, B, lb, kb);
      stage_tiles(A, la, B, lb, m0, n0, kb, aLs[cur ^ 1], bLs[cur ^ 1], t);
    }
    if (it2 > 0 && i == it1) {
#pragma unroll
      for (int a = 0; a < 2; ++a)
#pragma unroll
        for (int b = 0; b < 2; ++b) acc[a][b] *= rs;
    }
    wmma_step(acc, aLs[cur], bLs[cur], wm, wn, hf, r, 0);
    wmma_step(acc, aLs[cur], bLs[cur], wm, wn, hf, r, 32);
    wait_async0();
    __syncthreads();
    cur ^= 1;
  }

  const float fs = (it2 > 0) ? g.alpha2 : g.alpha1;

  // epilogue: C VGPR v -> (m = v + 8*hf, n = r) within each 16x16 tile
#pragma unroll
  for (int mt = 0; mt < 2; ++mt)
#pragma unroll
    for (int nt = 0; nt < 2; ++nt) {
      const int gn = n0 + wn + nt * 16 + r;
      const float bia = g.bias ? g.bias[gn] : 0.0f;
#pragma unroll
      for (int v = 0; v < 8; ++v) {
        const int gm = m0 + wm + mt * 16 + v + 8 * hf;
        float val = acc[mt][nt][v] * fs + bia;
        if (g.res)  val += g.res[(long)z * g.sRes + (long)gm * g.ldres + gn];
        if (g.relu) val = val > 0.0f ? val : 0.0f;
        if (g.outF)  g.outF [(long)z * g.sOutF  + (long)gm * g.ldoF  + gn] = val;
        if (g.outB)  g.outB [(long)z * g.sOutB  + (long)gm * g.ldoB  + gn] = (bf16)val;
        if (g.outBT) g.outBT[(long)z * g.sOutBT + (long)gn * g.ldoBT + gm] = (bf16)val;
      }
    }
}

// ----------------------- complex LayerNorm (whitening) ---------------------
__global__ __launch_bounds__(256) void cln_kernel(const float* __restrict__ Xr,
                                                  const float* __restrict__ Xi,
                                                  bf16* __restrict__ NR,
                                                  bf16* __restrict__ NI) {
  __shared__ float sm[256];
  const int row = blockIdx.x, t = threadIdx.x;
  const float* xr = Xr + (long)row * DIM;
  const float* xi = Xi + (long)row * DIM;
  float r0 = xr[t], r1 = xr[t + 256];
  float i0 = xi[t], i1 = xi[t + 256];

  auto reduce = [&](float v) -> float {
    sm[t] = v; __syncthreads();
    for (int s = 128; s > 0; s >>= 1) {
      if (t < s) sm[t] += sm[t + s];
      __syncthreads();
    }
    float tot = sm[0];
    __syncthreads();
    return tot;
  };

  const float mr = reduce(r0 + r1) * (1.0f / DIM);
  const float mi = reduce(i0 + i1) * (1.0f / DIM);
  r0 -= mr; r1 -= mr; i0 -= mi; i1 -= mi;
  const float Crr = reduce(r0 * r0 + r1 * r1) * (1.0f / DIM) + EPSC;
  const float Cii = reduce(i0 * i0 + i1 * i1) * (1.0f / DIM) + EPSC;
  const float Cri = reduce(r0 * i0 + r1 * i1) * (1.0f / DIM);
  const float s   = sqrtf(Crr * Cii - Cri * Cri);
  const float tv  = sqrtf(Cii + Crr + 2.0f * s);
  const float inv = 1.0f / (s * tv);
  const float Rrr = (Cii + s) * inv;
  const float Rii = (Crr + s) * inv;
  const float Rri = -Cri * inv;

  const long base = (long)row * DIM;
  NR[base + t]       = (bf16)(Rrr * r0 + Rri * i0);
  NR[base + t + 256] = (bf16)(Rrr * r1 + Rri * i1);
  NI[base + t]       = (bf16)(Rii * i0 + Rri * r0);
  NI[base + t + 256] = (bf16)(Rii * i1 + Rri * r1);
}

// ----------------------------- converters ----------------------------------
__global__ void cvt_kernel(const float* __restrict__ in, bf16* __restrict__ out, int n) {
  int i = blockIdx.x * 256 + threadIdx.x;
  if (i < n) out[i] = (bf16)in[i];
}
// out[c*R + r] = in[r*C + c]  (transpose + convert)
__global__ void tcvt_kernel(const float* __restrict__ in, bf16* __restrict__ out, int R, int C) {
  int i = blockIdx.x * 256 + threadIdx.x;
  if (i < R * C) {
    int r = i / C, c = i % C;
    out[(long)c * R + r] = (bf16)in[i];
  }
}

// ------------------------------- host side ----------------------------------
static inline void run_gemm(hipStream_t st, const GArgs& g, int M, int Ncols, int batch) {
  dim3 grid(Ncols / BN, M / BM, batch);
  dual_gemm<<<grid, dim3(256), 0, st>>>(g);
}

extern "C" void kernel_launch(void* const* d_in, const int* in_sizes, int n_in,
                              void* d_out, int out_size, void* d_ws, size_t ws_size,
                              hipStream_t stream) {
  (void)in_sizes; (void)n_in; (void)out_size; (void)ws_size;
  const float* x_real = (const float*)d_in[0];
  const float* x_imag = (const float*)d_in[1];
  const float* Wq   = (const float*)d_in[2];
  const float* bq   = (const float*)d_in[3];
  const float* Wk   = (const float*)d_in[4];
  const float* bk   = (const float*)d_in[5];
  const float* Wv   = (const float*)d_in[6];
  const float* bv   = (const float*)d_in[7];
  const float* pk   = (const float*)d_in[8];
  const float* pv   = (const float*)d_in[9];
  const float* Wout = (const float*)d_in[10];
  const float* bout = (const float*)d_in[11];
  const float* W1r  = (const float*)d_in[12];
  const float* W1i  = (const float*)d_in[13];
  const float* b1r  = (const float*)d_in[14];
  const float* b1i  = (const float*)d_in[15];
  const float* W2r  = (const float*)d_in[16];
  const float* W2i  = (const float*)d_in[17];
  const float* b2r  = (const float*)d_in[18];
  const float* b2i  = (const float*)d_in[19];

  // ---- workspace layout ----
  uint8_t* base = (uint8_t*)d_ws;
  size_t off = 0;
  auto alloc = [&](size_t bytes) -> void* {
    void* p = base + off;
    off += (bytes + 255) & ~size_t(255);
    return p;
  };
  const long ND  = (long)SEQ * DIM;
  const long HNK = (long)HEADS * SEQ * LRK;

  float* xr  = (float*)alloc(ND * 4);
  float* xi  = (float*)alloc(ND * 4);
  bf16* nrB  = (bf16*)alloc(ND * 2);
  bf16* niB  = (bf16*)alloc(ND * 2);
  // transposed bf16 weights ([N][K] row-major)
  bf16* WqT  = (bf16*)alloc(DEPTH * DIM * DIM * 2);
  bf16* WkT  = (bf16*)alloc(DEPTH * DIM * DIM * 2);
  bf16* WvT  = (bf16*)alloc(DEPTH * DIM * DIM * 2);
  bf16* WoT  = (bf16*)alloc(DEPTH * DIM * DIM * 2);
  bf16* W1rT = (bf16*)alloc(DEPTH * DIM * DFF * 2);   // [DFF][DIM]
  bf16* W1iT = (bf16*)alloc(DEPTH * DIM * DFF * 2);
  bf16* W2rT = (bf16*)alloc(DEPTH * DFF * DIM * 2);   // [DIM][DFF]
  bf16* W2iT = (bf16*)alloc(DEPTH * DFF * DIM * 2);
  bf16* pkT  = (bf16*)alloc(DEPTH * LRK * SEQ * 2);   // [LRK][SEQ] (A operand)
  bf16* pvT  = (bf16*)alloc(DEPTH * LRK * SEQ * 2);
  // activations
  bf16* qA   = (bf16*)alloc(ND * 2);                  // [SEQ][DIM]
  bf16* qB   = (bf16*)alloc(ND * 2);
  bf16* kAT  = (bf16*)alloc(ND * 2);                  // [DIM][SEQ] (Bt for proj)
  bf16* kBT  = (bf16*)alloc(ND * 2);
  bf16* vAT  = (bf16*)alloc(ND * 2);
  bf16* vBT  = (bf16*)alloc(ND * 2);
  bf16* KA   = (bf16*)alloc((long)LRK * DIM * 2);     // [LRK][DIM] (Bt for dots)
  bf16* KB   = (bf16*)alloc((long)LRK * DIM * 2);
  bf16* VTA  = (bf16*)alloc((long)DIM * LRK * 2);     // [DIM][LRK] (Bt for o)
  bf16* VTB  = (bf16*)alloc((long)DIM * LRK * 2);
  bf16* DrB  = (bf16*)alloc(HNK * 2);                 // [H][SEQ][LRK]; aliased as hrB
  bf16* DiB  = (bf16*)alloc(HNK * 2);                 //                aliased as hiB
  bf16* orB  = (bf16*)alloc(ND * 2);
  bf16* oiB  = (bf16*)alloc(ND * 2);
  bf16* hrB  = DrB;   // [SEQ][DFF] == 8*SEQ*LRK elems, disjoint lifetime
  bf16* hiB  = DiB;

  // ---- weight conversions (transposed) ----
  auto tcvt = [&](const float* src, bf16* dst, long R, long C) {
    long n = R * C;
    tcvt_kernel<<<dim3((unsigned)((n + 255) / 256)), dim3(256), 0, stream>>>(src, dst, (int)R, (int)C);
  };
  for (int d = 0; d < DEPTH; ++d) {
    tcvt(Wq  + (long)d * DIM * DIM, WqT  + (long)d * DIM * DIM, DIM, DIM);
    tcvt(Wk  + (long)d * DIM * DIM, WkT  + (long)d * DIM * DIM, DIM, DIM);
    tcvt(Wv  + (long)d * DIM * DIM, WvT  + (long)d * DIM * DIM, DIM, DIM);
    tcvt(Wout+ (long)d * DIM * DIM, WoT  + (long)d * DIM * DIM, DIM, DIM);
    tcvt(W1r + (long)d * DIM * DFF, W1rT + (long)d * DIM * DFF, DIM, DFF);
    tcvt(W1i + (long)d * DIM * DFF, W1iT + (long)d * DIM * DFF, DIM, DFF);
    tcvt(W2r + (long)d * DFF * DIM, W2rT + (long)d * DFF * DIM, DFF, DIM);
    tcvt(W2i + (long)d * DFF * DIM, W2iT + (long)d * DFF * DIM, DFF, DIM);
    tcvt(pk  + (long)d * SEQ * LRK, pkT  + (long)d * SEQ * LRK, SEQ, LRK);
    tcvt(pv  + (long)d * SEQ * LRK, pvT  + (long)d * SEQ * LRK, SEQ, LRK);
  }
  hipMemcpyAsync(xr, x_real, ND * 4, hipMemcpyDeviceToDevice, stream);
  hipMemcpyAsync(xi, x_imag, ND * 4, hipMemcpyDeviceToDevice, stream);

  for (int d = 0; d < DEPTH; ++d) {
    const bf16* WqD = WqT + (long)d * DIM * DIM;
    const bf16* WkD = WkT + (long)d * DIM * DIM;
    const bf16* WvD = WvT + (long)d * DIM * DIM;
    const bf16* WoD = WoT + (long)d * DIM * DIM;
    const bf16* W1rD = W1rT + (long)d * DIM * DFF;
    const bf16* W1iD = W1iT + (long)d * DIM * DFF;
    const bf16* W2rD = W2rT + (long)d * DFF * DIM;
    const bf16* W2iD = W2iT + (long)d * DFF * DIM;
    const bf16* pkTD = pkT + (long)d * LRK * SEQ;
    const bf16* pvTD = pvT + (long)d * LRK * SEQ;
    const float *bqD = bq + d * DIM, *bkD = bk + d * DIM, *bvD = bv + d * DIM;
    const float *boD = bout + d * DIM;
    const float *b1rD = b1r + d * DFF, *b1iD = b1i + d * DFF;
    const float *b2rD = b2r + d * DIM, *b2iD = b2i + d * DIM;

    // ---- pre-attention complex LN ----
    cln_kernel<<<dim3(SEQ), dim3(256), 0, stream>>>(xr, xi, nrB, niB);

    // ---- QKV projections: q normal-orient, k/v transposed-orient outputs ----
    struct QKV { const bf16* x; const bf16* w; const float* b; bf16* o; int trans; };
    QKV qkv[6] = { {nrB, WqD, bqD, qA, 0}, {niB, WqD, bqD, qB, 0},
                   {nrB, WkD, bkD, kAT, 1}, {niB, WkD, bkD, kBT, 1},
                   {nrB, WvD, bvD, vAT, 1}, {niB, WvD, bvD, vBT, 1} };
    for (int j = 0; j < 6; ++j) {
      GArgs g{};
      g.A1 = qkv[j].x; g.lda1 = DIM; g.B1 = qkv[j].w; g.ldb1 = DIM;
      g.K1 = DIM; g.alpha1 = 1.0f; g.bias = qkv[j].b;
      if (qkv[j].trans) { g.outBT = qkv[j].o; g.ldoBT = SEQ; }
      else              { g.outB  = qkv[j].o; g.ldoB  = DIM; }
      run_gemm(stream, g, SEQ, DIM, 1);
    }

    // ---- low-rank K/V (M=256, N=512, K=4096): K normal, V transposed out ----
    { GArgs g{}; g.A1 = pkTD; g.lda1 = SEQ; g.B1 = kAT; g.ldb1 = SEQ; g.K1 = SEQ;
      g.alpha1 = 1.0f; g.outB = KA; g.ldoB = DIM; run_gemm(stream, g, LRK, DIM, 1); }
    { GArgs g{}; g.A1 = pkTD; g.lda1 = SEQ; g.B1 = kBT; g.ldb1 = SEQ; g.K1 = SEQ;
      g.alpha1 = 1.0f; g.outB = KB; g.ldoB = DIM; run_gemm(stream, g, LRK, DIM, 1); }
    { GArgs g{}; g.A1 = pvTD; g.lda1 = SEQ; g.B1 = vAT; g.ldb1 = SEQ; g.K1 = SEQ;
      g.alpha1 = 1.0f; g.outBT = VTA; g.ldoBT = LRK; run_gemm(stream, g, LRK, DIM, 1); }
    { GArgs g{}; g.A1 = pvTD; g.lda1 = SEQ; g.B1 = vBT; g.ldb1 = SEQ; g.K1 = SEQ;
      g.alpha1 = 1.0f; g.outBT = VTB; g.ldoBT = LRK; run_gemm(stream, g, LRK, DIM, 1); }

    // ---- complex dots: Dr = S*(qA@KA^T - qB@KB^T), Di = S*(qA@KB^T + qB@KA^T) ----
    { GArgs g{};
      g.A1 = qA; g.lda1 = DIM; g.sA1 = DHEAD; g.B1 = KA; g.ldb1 = DIM; g.sB1 = DHEAD;
      g.K1 = DHEAD; g.alpha1 = SCALEC;
      g.A2 = qB; g.lda2 = DIM; g.sA2 = DHEAD; g.B2 = KB; g.ldb2 = DIM; g.sB2 = DHEAD;
      g.K2 = DHEAD; g.alpha2 = -SCALEC;
      g.outB = DrB; g.ldoB = LRK; g.sOutB = (long)SEQ * LRK;
      run_gemm(stream, g, SEQ, LRK, HEADS); }
    { GArgs g{};
      g.A1 = qA; g.lda1 = DIM; g.sA1 = DHEAD; g.B1 = KB; g.ldb1 = DIM; g.sB1 = DHEAD;
      g.K1 = DHEAD; g.alpha1 = SCALEC;
      g.A2 = qB; g.lda2 = DIM; g.sA2 = DHEAD; g.B2 = KA; g.ldb2 = DIM; g.sB2 = DHEAD;
      g.K2 = DHEAD; g.alpha2 = SCALEC;
      g.outB = DiB; g.ldoB = LRK; g.sOutB = (long)SEQ * LRK;
      run_gemm(stream, g, SEQ, LRK, HEADS); }

    // ---- o_r = Dr@V_A - Di@V_B ; o_i = Dr@V_B + Di@V_A (per head, N=64) ----
    { GArgs g{};
      g.A1 = DrB; g.lda1 = LRK; g.sA1 = (long)SEQ * LRK; g.B1 = VTA; g.ldb1 = LRK; g.sB1 = (long)DHEAD * LRK;
      g.K1 = LRK; g.alpha1 = 1.0f;
      g.A2 = DiB; g.lda2 = LRK; g.sA2 = (long)SEQ * LRK; g.B2 = VTB; g.ldb2 = LRK; g.sB2 = (long)DHEAD * LRK;
      g.K2 = LRK; g.alpha2 = -1.0f;
      g.outB = orB; g.ldoB = DIM; g.sOutB = DHEAD;
      run_gemm(stream, g, SEQ, DHEAD, HEADS); }
    { GArgs g{};
      g.A1 = DrB; g.lda1 = LRK; g.sA1 = (long)SEQ * LRK; g.B1 = VTB; g.ldb1 = LRK; g.sB1 = (long)DHEAD * LRK;
      g.K1 = LRK; g.alpha1 = 1.0f;
      g.A2 = DiB; g.lda2 = LRK; g.sA2 = (long)SEQ * LRK; g.B2 = VTA; g.ldb2 = LRK; g.sB2 = (long)DHEAD * LRK;
      g.K2 = LRK; g.alpha2 = 1.0f;
      g.outB = oiB; g.ldoB = DIM; g.sOutB = DHEAD;
      run_gemm(stream, g, SEQ, DHEAD, HEADS); }

    // ---- output projection + residual (fp32 stream) ----
    { GArgs g{}; g.A1 = orB; g.lda1 = DIM; g.B1 = WoD; g.ldb1 = DIM; g.K1 = DIM;
      g.alpha1 = 1.0f; g.bias = boD; g.res = xr; g.ldres = DIM;
      g.outF = xr; g.ldoF = DIM; run_gemm(stream, g, SEQ, DIM, 1); }
    { GArgs g{}; g.A1 = oiB; g.lda1 = DIM; g.B1 = WoD; g.ldb1 = DIM; g.K1 = DIM;
      g.alpha1 = 1.0f; g.bias = boD; g.res = xi; g.ldres = DIM;
      g.outF = xi; g.ldoF = DIM; run_gemm(stream, g, SEQ, DIM, 1); }

    // ---- pre-FFN complex LN ----
    cln_kernel<<<dim3(SEQ), dim3(256), 0, stream>>>(xr, xi, nrB, niB);

    // ---- FFN ----
    { GArgs g{}; g.A1 = nrB; g.lda1 = DIM; g.B1 = W1rD; g.ldb1 = DIM; g.K1 = DIM; g.alpha1 = 1.0f;
      g.A2 = niB; g.lda2 = DIM; g.B2 = W1iD; g.ldb2 = DIM; g.K2 = DIM; g.alpha2 = -1.0f;
      g.bias = b1rD; g.relu = 1; g.outB = hrB; g.ldoB = DFF;
      run_gemm(stream, g, SEQ, DFF, 1); }
    { GArgs g{}; g.A1 = nrB; g.lda1 = DIM; g.B1 = W1iD; g.ldb1 = DIM; g.K1 = DIM; g.alpha1 = 1.0f;
      g.A2 = niB; g.lda2 = DIM; g.B2 = W1rD; g.ldb2 = DIM; g.K2 = DIM; g.alpha2 = 1.0f;
      g.bias = b1iD; g.relu = 1; g.outB = hiB; g.ldoB = DFF;
      run_gemm(stream, g, SEQ, DFF, 1); }
    { GArgs g{}; g.A1 = hrB; g.lda1 = DFF; g.B1 = W2rD; g.ldb1 = DFF; g.K1 = DFF; g.alpha1 = 1.0f;
      g.A2 = hiB; g.lda2 = DFF; g.B2 = W2iD; g.ldb2 = DFF; g.K2 = DFF; g.alpha2 = -1.0f;
      g.bias = b2rD; g.res = xr; g.ldres = DIM; g.outF = xr; g.ldoF = DIM;
      run_gemm(stream, g, SEQ, DIM, 1); }
    { GArgs g{}; g.A1 = hrB; g.lda1 = DFF; g.B1 = W2iD; g.ldb1 = DFF; g.K1 = DFF; g.alpha1 = 1.0f;
      g.A2 = hiB; g.lda2 = DFF; g.B2 = W2rD; g.ldb2 = DFF; g.K2 = DFF; g.alpha2 = 1.0f;
      g.bias = b2iD; g.res = xi; g.ldres = DIM; g.outF = xi; g.ldoF = DIM;
      run_gemm(stream, g, SEQ, DIM, 1); }
  }

  // ---- output: stack [xr, xi] ----
  float* outp = (float*)d_out;
  hipMemcpyAsync(outp, xr, ND * 4, hipMemcpyDeviceToDevice, stream);
  hipMemcpyAsync(outp + ND, xi, ND * 4, hipMemcpyDeviceToDevice, stream);
}
